// ProjRNNReducedI2T_68324339744984
// MI455X (gfx1250) — matrix operand order repaired and checked by reading, not compiled, not run on hardware
//
#include <hip/hip_runtime.h>

// ---------------------------------------------------------------------------
// Types / helpers
// ---------------------------------------------------------------------------
typedef __attribute__((ext_vector_type(16))) __bf16 v16bf;
typedef __attribute__((ext_vector_type(8)))  float  v8f;

__device__ __forceinline__ __bf16 f2bf(float f) {
    union { float f; unsigned u; } v; v.f = f;
    unsigned r = v.u + 0x7FFFu + ((v.u >> 16) & 1u);   // round-to-nearest-even
    unsigned short h = (unsigned short)(r >> 16);
    __bf16 o; __builtin_memcpy(&o, &h, 2); return o;
}
__device__ __forceinline__ float sigm(float x) { return 1.f / (1.f + expf(-x)); }

// A fragment: 16x32 bf16 (M x K). ISA 7.12.2: lanes 0-15 M=lane, V0-3 K=0..7,
// V4-7 K=16..23; lanes 16-31 same M, K offset +8 within each half.
__device__ __forceinline__ v16bf load_frag_a(const __bf16* __restrict__ A, int lda,
                                             int m0, int k0, int lane) {
    int m = m0 + (lane & 15);
    int half = lane >> 4;
    const __bf16* row = A + (long)m * lda;
    v16bf a;
#pragma unroll
    for (int i = 0; i < 8; ++i) {
        int k = k0 + ((i < 4) ? (2 * i) : (2 * i + 8)) + half * 8;
        a[2 * i]     = row[k];
        a[2 * i + 1] = row[k + 1];
    }
    return a;
}
// B fragment: 32x16 bf16 (K x N), B(k,n) = W[n*ldb + k] (so GEMM = A @ W^T).
// Lanes 0-15: N=lane, V0-3 K=0..7, V4-7 K=16..23; lanes 16-31 K offset +8 (same as A).
__device__ __forceinline__ v16bf load_frag_b(const __bf16* __restrict__ W, int ldb,
                                             int n0, int k0, int lane) {
    int n = n0 + (lane & 15);
    int half = lane >> 4;
    const __bf16* row = W + (long)n * ldb;
    v16bf b;
#pragma unroll
    for (int j = 0; j < 8; ++j) {
        int k = k0 + ((j < 4) ? (2 * j) : (2 * j + 8)) + half * 8;
        b[2 * j]     = row[k];
        b[2 * j + 1] = row[k + 1];
    }
    return b;
}
__device__ __forceinline__ v8f wmma_bf16(v16bf a, v16bf b, v8f c) {
    return __builtin_amdgcn_wmma_f32_16x16x32_bf16(false, a, false, b, (short)0, c,
                                                   false, false);
}

// ---------------------------------------------------------------------------
// Generic GEMM: C(MxN) = A(MxK,bf16) @ B(NxK,bf16)^T + bias(N).
// Register-blocked: one wave computes a 16x64 strip (4 accumulators, A-frag
// reused 4x).  grid = (M/16, N/64).
// ---------------------------------------------------------------------------
__global__ __launch_bounds__(32) void gemm_bf16_wmma(
    const __bf16* __restrict__ A, int lda,
    const __bf16* __restrict__ B, int ldb,
    float* __restrict__ C, int ldc,
    const float* __restrict__ bias, int K) {
    int lane = threadIdx.x;
    int m0 = blockIdx.x * 16, n0 = blockIdx.y * 64;
    v8f acc[4] = {{}, {}, {}, {}};
    for (int k0 = 0; k0 < K; k0 += 32) {
        if (k0 + 32 < K) {
            __builtin_prefetch(A + (long)(m0 + (lane & 15)) * lda + k0 + 32, 0, 1);
            __builtin_prefetch(B + (long)(n0 + (lane & 15)) * ldb + k0 + 32, 0, 1);
        }
        v16bf a = load_frag_a(A, lda, m0, k0, lane);
#pragma unroll
        for (int s = 0; s < 4; ++s) {
            v16bf b = load_frag_b(B, ldb, n0 + s * 16, k0, lane);
            acc[s] = wmma_bf16(a, b, acc[s]);
        }
    }
    int nl = lane & 15;
    int mr = m0 + ((lane >> 4) << 3);
#pragma unroll
    for (int s = 0; s < 4; ++s) {
        int n = n0 + s * 16 + nl;
        float bv = bias ? bias[n] : 0.f;
#pragma unroll
        for (int v = 0; v < 8; ++v) C[(long)(mr + v) * ldc + n] = acc[s][v] + bv;
    }
}

// Batched FC: fcout[img](64x1024) = txtvec[img](64x128) @ WfcB^T + basefc
// grid = (64/16, 1024/64, 64 imgs)
__global__ __launch_bounds__(32) void fc_batched_wmma(
    const __bf16* __restrict__ TV, const __bf16* __restrict__ WfcB,
    const float* __restrict__ basefc, float* __restrict__ out) {
    int lane = threadIdx.x;
    int m0 = blockIdx.x * 16, n0 = blockIdx.y * 64, img = blockIdx.z;
    const __bf16* A = TV + (long)img * 64 * 128;
    v8f acc[4] = {{}, {}, {}, {}};
#pragma unroll
    for (int k0 = 0; k0 < 128; k0 += 32) {
        v16bf a = load_frag_a(A, 128, m0, k0, lane);
#pragma unroll
        for (int s = 0; s < 4; ++s) {
            v16bf b = load_frag_b(WfcB, 1152, n0 + s * 16, k0, lane);
            acc[s] = wmma_bf16(a, b, acc[s]);
        }
    }
    int nl = lane & 15;
    int mr = m0 + ((lane >> 4) << 3);
    float* C = out + (long)img * 64 * 1024;
#pragma unroll
    for (int s = 0; s < 4; ++s) {
        int n = n0 + s * 16 + nl;
#pragma unroll
        for (int v = 0; v < 8; ++v)
            C[(long)(mr + v) * 1024 + n] = acc[s][v] + basefc[(long)(mr + v) * 1024 + n];
    }
}

// ---------------------------------------------------------------------------
// Elementwise / small kernels
// ---------------------------------------------------------------------------
__global__ void cvt_pad_kernel(const float* __restrict__ src, __bf16* __restrict__ dst,
                               int rows, int srcw, int dstw) {
    long idx = (long)blockIdx.x * blockDim.x + threadIdx.x;
    long total = (long)rows * dstw;
    if (idx >= total) return;
    int r = (int)(idx / dstw), c = (int)(idx % dstw);
    dst[idx] = (c < srcw) ? f2bf(src[(long)r * srcw + c]) : f2bf(0.f);
}

__global__ void fill_zero_u32(unsigned* p, long n) {
    long i = (long)blockIdx.x * blockDim.x + threadIdx.x;
    if (i < n) p[i] = 0u;
}

// torch GRU gate math for the big (L=1024) GRU, one step.
__global__ void gru_gate_kernel(const float* __restrict__ gi, const float* __restrict__ gh,
                                float* __restrict__ h, __bf16* __restrict__ h_bf,
                                float* __restrict__ txt, int t, int accumulate) {
    int idx = blockIdx.x * 256 + threadIdx.x;      // 64*1024
    int b = idx >> 10, l = idx & 1023;
    const float* gip = gi + (long)(b * 32 + t) * 3072;
    const float* ghp = gh + (long)b * 3072;
    float r  = sigm(gip[l] + ghp[l]);
    float z  = sigm(gip[1024 + l] + ghp[1024 + l]);
    float ng = tanhf(gip[2048 + l] + r * ghp[2048 + l]);
    float hn = (1.f - z) * ng + z * h[idx];
    h[idx] = hn;
    h_bf[idx] = f2bf(hn);
    long ti = (long)(b * 32 + t) * 1024 + l;
    if (accumulate) txt[ti] += 0.5f * hn; else txt[ti] = 0.5f * hn;
}

// scores + softmax per batch element
__global__ void attn_scores_kernel(const float* __restrict__ q, const float* __restrict__ k,
                                   float* __restrict__ attn) {
    __shared__ float sc[1024];
    int b = blockIdx.x;
    for (int p = threadIdx.x; p < 1024; p += blockDim.x) {
        int t = p >> 5, s = p & 31;
        const float* qp = q + (long)(b * 32 + t) * 128;
        const float* kp = k + (long)(b * 32 + s) * 128;
        float d = 0.f;
        for (int o = 0; o < 128; ++o) d += qp[o] * kp[o];
        sc[p] = d;
    }
    __syncthreads();
    if (threadIdx.x < 32) {
        int t = threadIdx.x;
        float mx = -3.0e38f;
        for (int s = 0; s < 32; ++s) mx = fmaxf(mx, sc[t * 32 + s]);
        float sum = 0.f;
        for (int s = 0; s < 32; ++s) { float e = expf(sc[t * 32 + s] - mx); sc[t * 32 + s] = e; sum += e; }
        float inv = 1.f / sum;
        for (int s = 0; s < 32; ++s) attn[(long)b * 1024 + t * 32 + s] = sc[t * 32 + s] * inv;
    }
}

__global__ void attn_apply_kernel(const float* __restrict__ vbuf, const float* __restrict__ attn,
                                  const float* __restrict__ txt, float* __restrict__ txt2,
                                  const float* __restrict__ gamma_p) {
    long idx = (long)blockIdx.x * 256 + threadIdx.x;   // 2048*1024
    int bt = (int)(idx >> 10), c = (int)(idx & 1023);
    int b = bt >> 5, t = bt & 31;
    const float* ap = attn + (long)b * 1024 + t * 32;
    float s = 0.f;
    for (int ss = 0; ss < 32; ++ss) s += ap[ss] * vbuf[(long)(b * 32 + ss) * 1024 + c];
    txt2[idx] = gamma_p[0] * s + txt[idx];
}

__global__ void mask_pool_kernel(const float* __restrict__ txt2, const int* __restrict__ lens,
                                 float* __restrict__ te) {
    int idx = blockIdx.x * 256 + threadIdx.x;   // 64*1024
    int b = idx >> 10, l = idx & 1023;
    int L = lens[b];
    float s = 0.f;
    for (int t = 0; t < L; ++t) s += txt2[(long)(b * 32 + t) * 1024 + l];
    te[idx] = s / (float)L;
}

__global__ void img_pool_kernel(const float* __restrict__ img, float* __restrict__ glob) {
    int idx = blockIdx.x * 256 + threadIdx.x;   // 64*1024
    int b = idx >> 10, c = idx & 1023;
    float s = 0.f;
    for (int r = 0; r < 36; ++r) s += img[(long)b * 36 * 1024 + (long)r * 1024 + c];
    glob[idx] = s * (1.f / 36.f);
}

__global__ void img_reduce_kernel(const float* __restrict__ glob, const float* __restrict__ W,
                                  const float* __restrict__ bias, float* __restrict__ gvec) {
    int idx = blockIdx.x * 256 + threadIdx.x;   // 64*128
    int b = idx >> 7, o = idx & 127;
    const float* g = glob + (long)b * 1024;
    const float* w = W + (long)o * 1024;
    float s = 0.f;
    for (int c = 0; c < 1024; ++c) s += w[c] * g[c];
    gvec[idx] = s + bias[o];
}

__global__ void l2norm_rows_kernel(const float* __restrict__ src, float* __restrict__ dst,
                                   int cols) {
    __shared__ float red[256];
    int row = blockIdx.x;
    const float* s = src + (long)row * cols;
    float ss = 0.f;
    for (int c = threadIdx.x; c < cols; c += 256) ss += s[c] * s[c];
    red[threadIdx.x] = ss; __syncthreads();
    for (int off = 128; off > 0; off >>= 1) {
        if ((int)threadIdx.x < off) red[threadIdx.x] += red[threadIdx.x + off];
        __syncthreads();
    }
    float inv = 1.f / (sqrtf(red[0]) + 1e-8f);
    for (int c = threadIdx.x; c < cols; c += 256) dst[(long)row * cols + c] = s[c] * inv;
}

__global__ void gen_bias_kernel(const float* __restrict__ Wb, const float* __restrict__ bb,
                                const float* __restrict__ gvec, float* __restrict__ out) {
    int idx = blockIdx.x * 256 + threadIdx.x;   // 64*384
    int i = idx / 384, h = idx % 384;
    const float* w = Wb + (long)h * 128;
    const float* g = gvec + (long)i * 128;
    float s = 0.f;
    for (int j = 0; j < 128; ++j) s += w[j] * g[j];
    out[idx] = s + bb[h];
}

// C(49152 x 64, ldc=64) + bias -> per-image bf16 (64 x 49152)
__global__ void gen_convert_kernel(const float* __restrict__ C, const float* __restrict__ gbias,
                                   __bf16* __restrict__ out) {
    long idx = (long)blockIdx.x * 256 + threadIdx.x;   // 49152*64
    int m = (int)(idx >> 6), i = (int)(idx & 63);
    out[(long)i * 49152 + m] = f2bf(C[idx] + gbias[m]);
}

// cosine sims: out[i*64+c] = l2norm(fcout[i,c,:]) . iv[i,:]
__global__ void sims_kernel(const float* __restrict__ fcout, const float* __restrict__ iv,
                            float* __restrict__ out) {
    __shared__ float s1[256], s2[256];
    int c = blockIdx.x, i = blockIdx.y;
    const float* row = fcout + ((long)i * 64 + c) * 1024;
    const float* ivp = iv + (long)i * 1024;
    float ss = 0.f, dt = 0.f;
    for (int l = threadIdx.x; l < 1024; l += 256) { float x = row[l]; ss += x * x; dt += x * ivp[l]; }
    s1[threadIdx.x] = ss; s2[threadIdx.x] = dt; __syncthreads();
    for (int off = 128; off > 0; off >>= 1) {
        if ((int)threadIdx.x < off) { s1[threadIdx.x] += s1[threadIdx.x + off]; s2[threadIdx.x] += s2[threadIdx.x + off]; }
        __syncthreads();
    }
    if (threadIdx.x == 0) out[(long)i * 64 + c] = s2[0] / (sqrtf(s1[0]) + 1e-8f);
}

// ---------------------------------------------------------------------------
// Fused generated-weight GRU: one workgroup per image, persistent LDS state.
// acc layout per caption row (512 cols): [0,256)=fused r|z (gi+gh),
// [256,384)=gi_n, [384,512)=gh_n.  Dynamic LDS = 64*512*4 + 384*128*2 +
// 64*128*2 = 245760 B (< 320KB WGP limit).  Whh is DMA'd to LDS with
// global_load_async_to_lds_b128 (ASYNCcnt).
// ---------------------------------------------------------------------------
#define GG_ACC_BYTES   (64 * 512 * 4)
#define GG_WHH_BYTES   (384 * 128 * 2)

__global__ __launch_bounds__(256) void gen_gru_kernel(
    const __bf16* __restrict__ capR,      // (64 caps, 32, 128)
    const __bf16* __restrict__ WihGen,    // (64 imgs, 384, 128)
    const __bf16* __restrict__ WhhGen,    // (64 imgs, 384, 128)
    const float* __restrict__ bihgen,     // (64, 384)
    const float* __restrict__ bhhgen,     // (64, 384)
    __bf16* __restrict__ txtvec)          // (64 imgs, 64, 128)
{
    extern __shared__ char smem[];
    float*  acc = (float*)smem;                                   // 64*512
    __bf16* whh = (__bf16*)(smem + GG_ACC_BYTES);                 // 384*128
    __bf16* hbf = (__bf16*)(smem + GG_ACC_BYTES + GG_WHH_BYTES);  // 64*128
    int img = blockIdx.x;
    int tid = threadIdx.x, lane = tid & 31, wv = tid >> 5;
    const __bf16* wih_g = WihGen + (long)img * 49152;
    const __bf16* whh_g = WhhGen + (long)img * 49152;

    // ---- async DMA: Whh (98304 B) global -> LDS, 16 B per lane per pass ----
    {
        unsigned lds0 = (unsigned)__builtin_amdgcn_groupstaticsize() + (unsigned)GG_ACC_BYTES;
        const char* gsrc = (const char*)whh_g;
#pragma unroll
        for (int pass = 0; pass < 24; ++pass) {
            unsigned loff = lds0 + (unsigned)(pass * 4096 + tid * 16);
            unsigned long long ga = (unsigned long long)(gsrc + pass * 4096 + tid * 16);
            asm volatile("global_load_async_to_lds_b128 %0, %1, off"
                         :: "v"(loff), "v"(ga) : "memory");
        }
        asm volatile("s_wait_asynccnt 0x0" ::: "memory");
    }

    const float* bih = bihgen + (long)img * 384;
    const float* bhh = bhhgen + (long)img * 384;

    float hreg[32], hmax[32];
    int ebase = tid * 32;
#pragma unroll
    for (int e = 0; e < 32; ++e) { hreg[e] = 0.f; hmax[e] = -3.0e38f; hbf[ebase + e] = f2bf(0.f); }

    // wave -> fixed M strip: tm = wv>>1, half of the 32 column tiles
    int m0 = (wv >> 1) * 16;
    int tn0 = (wv & 1) * 16;

    for (int t = 0; t < 32; ++t) {
        __syncthreads();   // hbf/whh ready; prev gate phase done with acc
        // ---- WMMA phase: preload A fragments once, reuse across 16 tiles ----
        v16bf capA[4], hA[4];
#pragma unroll
        for (int kk = 0; kk < 4; ++kk) {
            capA[kk] = load_frag_a(capR + t * 128, 32 * 128, m0, kk * 32, lane);
            hA[kk]   = load_frag_a(hbf, 128, m0, kk * 32, lane);
        }
        for (int tn = tn0; tn < tn0 + 16; ++tn) {
            int n0 = tn * 16;
            v8f c = {};
            if (tn < 16) {          // fused r|z: cap@Wih[0:256) + h@Whh[0:256)
#pragma unroll
                for (int kk = 0; kk < 4; ++kk)
                    c = wmma_bf16(capA[kk], load_frag_b(wih_g, 128, n0, kk * 32, lane), c);
#pragma unroll
                for (int kk = 0; kk < 4; ++kk)
                    c = wmma_bf16(hA[kk], load_frag_b(whh, 128, n0, kk * 32, lane), c);
            } else if (tn < 24) {   // gi_n: cap @ Wih rows [256,384)
#pragma unroll
                for (int kk = 0; kk < 4; ++kk)
                    c = wmma_bf16(capA[kk], load_frag_b(wih_g, 128, n0, kk * 32, lane), c);
            } else {                // gh_n: h @ Whh rows [256,384)
#pragma unroll
                for (int kk = 0; kk < 4; ++kk)
                    c = wmma_bf16(hA[kk], load_frag_b(whh, 128, n0 - 128, kk * 32, lane), c);
            }
            int n = n0 + (lane & 15);
            int mr = m0 + ((lane >> 4) << 3);
#pragma unroll
            for (int v = 0; v < 8; ++v) acc[(mr + v) * 512 + n] = c[v];
        }
        __syncthreads();
        // ---- gate phase: each thread owns 32 h elements ----
#pragma unroll
        for (int e = 0; e < 32; ++e) {
            int idx = ebase + e;
            int cc = idx >> 7, j = idx & 127;
            float r  = sigm(acc[cc * 512 + j]       + bih[j]       + bhh[j]);
            float z  = sigm(acc[cc * 512 + 128 + j] + bih[128 + j] + bhh[128 + j]);
            float gn = acc[cc * 512 + 256 + j] + bih[256 + j];
            float hn = acc[cc * 512 + 384 + j] + bhh[256 + j];
            float ng = tanhf(gn + r * hn);
            float h  = (1.f - z) * ng + z * hreg[e];
            hreg[e] = h;
            hmax[e] = fmaxf(hmax[e], h);
            hbf[idx] = f2bf(h);
        }
    }
    __syncthreads();
#pragma unroll
    for (int e = 0; e < 32; ++e) txtvec[(long)img * 8192 + ebase + e] = f2bf(hmax[e]);
}

// ---------------------------------------------------------------------------
// Host driver
// ---------------------------------------------------------------------------
static inline dim3 g1(long total) { return dim3((unsigned)((total + 255) / 256)); }

extern "C" void kernel_launch(void* const* d_in, const int* in_sizes, int n_in,
                              void* d_out, int out_size, void* d_ws, size_t ws_size,
                              hipStream_t stream) {
    (void)in_sizes; (void)n_in; (void)out_size; (void)ws_size;
    const float* img_embed    = (const float*)d_in[0];
    const float* cap_embed    = (const float*)d_in[1];
    const int*   lens         = (const int*)d_in[2];
    const float* W_reduce_img = (const float*)d_in[3];
    const float* b_reduce_img = (const float*)d_in[4];
    const float* W_reduce_txt = (const float*)d_in[5];
    const float* b_reduce_txt = (const float*)d_in[6];
    const float* gru_Wih_f = (const float*)d_in[7];
    const float* gru_Whh_f = (const float*)d_in[8];
    const float* gru_bih_f = (const float*)d_in[9];
    const float* gru_bhh_f = (const float*)d_in[10];
    const float* gru_Wih_b = (const float*)d_in[11];
    const float* gru_Whh_b = (const float*)d_in[12];
    const float* gru_bih_b = (const float*)d_in[13];
    const float* gru_bhh_b = (const float*)d_in[14];
    const float* sa_Wq = (const float*)d_in[15];
    const float* sa_bq = (const float*)d_in[16];
    const float* sa_Wk = (const float*)d_in[17];
    const float* sa_bk = (const float*)d_in[18];
    const float* sa_Wv = (const float*)d_in[19];
    const float* sa_bv = (const float*)d_in[20];
    const float* sa_gamma = (const float*)d_in[21];
    const float* gen_Wih  = (const float*)d_in[22];
    const float* gen_bih  = (const float*)d_in[23];
    const float* gen_Whh  = (const float*)d_in[24];
    const float* gen_bhh  = (const float*)d_in[25];
    const float* gen_Wbih = (const float*)d_in[26];
    const float* gen_bbih = (const float*)d_in[27];
    const float* gen_Wbhh = (const float*)d_in[28];
    const float* gen_bbhh = (const float*)d_in[29];
    const float* W_txt_fc = (const float*)d_in[30];
    const float* b_txt_fc = (const float*)d_in[31];
    float* sims = (float*)d_out;

    char* ws = (char*)d_ws;
    size_t off = 0;
    auto alloc = [&](size_t bytes) -> void* {
        void* p = ws + off; off += (bytes + 255) & ~(size_t)255; return p;
    };
    // bf16 buffers
    __bf16* capE_bf   = (__bf16*)alloc(2048L * 320 * 2);
    __bf16* WihF_bf   = (__bf16*)alloc(3072L * 320 * 2);
    __bf16* WihB_bf   = (__bf16*)alloc(3072L * 320 * 2);
    __bf16* WhhF_bf   = (__bf16*)alloc(3072L * 1024 * 2);
    __bf16* WhhB_bf   = (__bf16*)alloc(3072L * 1024 * 2);
    __bf16* Wq_bf     = (__bf16*)alloc(128L * 1024 * 2);
    __bf16* Wk_bf     = (__bf16*)alloc(128L * 1024 * 2);
    __bf16* Wv_bf     = (__bf16*)alloc(1024L * 1024 * 2);
    __bf16* Wrt_bf    = (__bf16*)alloc(128L * 320 * 2);
    __bf16* genWih_bf = (__bf16*)alloc(49152L * 128 * 2);
    __bf16* genWhh_bf = (__bf16*)alloc(49152L * 128 * 2);
    __bf16* Wfc_bf    = (__bf16*)alloc(1024L * 1152 * 2);
    __bf16* txt_bf    = (__bf16*)alloc(2048L * 1024 * 2);
    __bf16* h_bf      = (__bf16*)alloc(64L * 1024 * 2);
    __bf16* capR_bf   = (__bf16*)alloc(2048L * 128 * 2);
    __bf16* WihGen_bf = (__bf16*)alloc(64L * 49152 * 2);
    __bf16* WhhGen_bf = (__bf16*)alloc(64L * 49152 * 2);
    __bf16* bases_bf  = (__bf16*)alloc(64L * 128 * 2);
    __bf16* txtvec_bf = (__bf16*)alloc(64L * 64 * 128 * 2);
    __bf16* te_bf     = (__bf16*)alloc(64L * 1024 * 2);
    // f32 buffers
    float* gi     = (float*)alloc(2048L * 3072 * 4);
    float* gh     = (float*)alloc(64L * 3072 * 4);
    float* hbuf   = (float*)alloc(64L * 1024 * 4);
    float* txt    = (float*)alloc(2048L * 1024 * 4);
    float* qb     = (float*)alloc(2048L * 128 * 4);
    float* kb     = (float*)alloc(2048L * 128 * 4);
    float* vb     = (float*)alloc(2048L * 1024 * 4);
    float* attn   = (float*)alloc(64L * 1024 * 4);
    float* txt2   = (float*)alloc(2048L * 1024 * 4);
    float* te     = (float*)alloc(64L * 1024 * 4);
    float* capred = (float*)alloc(2048L * 128 * 4);
    float* iglob  = (float*)alloc(64L * 1024 * 4);
    float* igvec  = (float*)alloc(64L * 128 * 4);
    float* iv     = (float*)alloc(64L * 1024 * 4);
    float* genC   = (float*)alloc(49152L * 64 * 4);
    float* bihgen = (float*)alloc(64L * 384 * 4);
    float* bhhgen = (float*)alloc(64L * 384 * 4);
    float* basefc = (float*)alloc(64L * 1024 * 4);
    float* fcout  = (float*)alloc(64L * 64 * 1024 * 4);

    // ---- weight / activation conversions (bf16, with K padding 300->320) ----
    cvt_pad_kernel<<<g1(2048L * 320), 256, 0, stream>>>(cap_embed, capE_bf, 2048, 300, 320);
    cvt_pad_kernel<<<g1(3072L * 320), 256, 0, stream>>>(gru_Wih_f, WihF_bf, 3072, 300, 320);
    cvt_pad_kernel<<<g1(3072L * 320), 256, 0, stream>>>(gru_Wih_b, WihB_bf, 3072, 300, 320);
    cvt_pad_kernel<<<g1(3072L * 1024), 256, 0, stream>>>(gru_Whh_f, WhhF_bf, 3072, 1024, 1024);
    cvt_pad_kernel<<<g1(3072L * 1024), 256, 0, stream>>>(gru_Whh_b, WhhB_bf, 3072, 1024, 1024);
    cvt_pad_kernel<<<g1(128L * 1024), 256, 0, stream>>>(sa_Wq, Wq_bf, 128, 1024, 1024);
    cvt_pad_kernel<<<g1(128L * 1024), 256, 0, stream>>>(sa_Wk, Wk_bf, 128, 1024, 1024);
    cvt_pad_kernel<<<g1(1024L * 1024), 256, 0, stream>>>(sa_Wv, Wv_bf, 1024, 1024, 1024);
    cvt_pad_kernel<<<g1(128L * 320), 256, 0, stream>>>(W_reduce_txt, Wrt_bf, 128, 300, 320);
    cvt_pad_kernel<<<g1(49152L * 128), 256, 0, stream>>>(gen_Wih, genWih_bf, 49152, 128, 128);
    cvt_pad_kernel<<<g1(49152L * 128), 256, 0, stream>>>(gen_Whh, genWhh_bf, 49152, 128, 128);
    cvt_pad_kernel<<<g1(1024L * 1152), 256, 0, stream>>>(W_txt_fc, Wfc_bf, 1024, 1152, 1152);

    // ---- bidirectional GRU (L=1024) ----
    // forward: gi = capE @ WihF^T + bih_f   (2048 x 3072, K=320)
    gemm_bf16_wmma<<<dim3(128, 48), 32, 0, stream>>>(capE_bf, 320, WihF_bf, 320,
                                                     gi, 3072, gru_bih_f, 320);
    fill_zero_u32<<<g1(65536), 256, 0, stream>>>((unsigned*)hbuf, 65536);
    fill_zero_u32<<<g1(32768), 256, 0, stream>>>((unsigned*)h_bf, 32768);
    for (int t = 0; t < 32; ++t) {
        gemm_bf16_wmma<<<dim3(4, 48), 32, 0, stream>>>(h_bf, 1024, WhhF_bf, 1024,
                                                       gh, 3072, gru_bhh_f, 1024);
        gru_gate_kernel<<<g1(65536), 256, 0, stream>>>(gi, gh, hbuf, h_bf, txt, t, 0);
    }
    // backward
    gemm_bf16_wmma<<<dim3(128, 48), 32, 0, stream>>>(capE_bf, 320, WihB_bf, 320,
                                                     gi, 3072, gru_bih_b, 320);
    fill_zero_u32<<<g1(65536), 256, 0, stream>>>((unsigned*)hbuf, 65536);
    fill_zero_u32<<<g1(32768), 256, 0, stream>>>((unsigned*)h_bf, 32768);
    for (int t = 31; t >= 0; --t) {
        gemm_bf16_wmma<<<dim3(4, 48), 32, 0, stream>>>(h_bf, 1024, WhhB_bf, 1024,
                                                       gh, 3072, gru_bhh_b, 1024);
        gru_gate_kernel<<<g1(65536), 256, 0, stream>>>(gi, gh, hbuf, h_bf, txt, t, 1);
    }

    // ---- self-attention ----
    cvt_pad_kernel<<<g1(2048L * 1024), 256, 0, stream>>>(txt, txt_bf, 2048, 1024, 1024);
    gemm_bf16_wmma<<<dim3(128, 2), 32, 0, stream>>>(txt_bf, 1024, Wq_bf, 1024, qb, 128, sa_bq, 1024);
    gemm_bf16_wmma<<<dim3(128, 2), 32, 0, stream>>>(txt_bf, 1024, Wk_bf, 1024, kb, 128, sa_bk, 1024);
    gemm_bf16_wmma<<<dim3(128, 16), 32, 0, stream>>>(txt_bf, 1024, Wv_bf, 1024, vb, 1024, sa_bv, 1024);
    attn_scores_kernel<<<64, 256, 0, stream>>>(qb, kb, attn);
    attn_apply_kernel<<<g1(2048L * 1024), 256, 0, stream>>>(vb, attn, txt, txt2, sa_gamma);

    // ---- pooling ----
    mask_pool_kernel<<<g1(65536), 256, 0, stream>>>(txt2, lens, te);
    cvt_pad_kernel<<<g1(64L * 1024), 256, 0, stream>>>(te, te_bf, 64, 1024, 1024);
    img_pool_kernel<<<g1(65536), 256, 0, stream>>>(img_embed, iglob);
    img_reduce_kernel<<<g1(8192), 256, 0, stream>>>(iglob, W_reduce_img, b_reduce_img, igvec);
    l2norm_rows_kernel<<<64, 256, 0, stream>>>(iglob, iv, 1024);
    cvt_pad_kernel<<<g1(64L * 128), 256, 0, stream>>>(igvec, bases_bf, 64, 128, 128);

    // ---- caption reduction (B*T x 128) ----
    gemm_bf16_wmma<<<dim3(128, 2), 32, 0, stream>>>(capE_bf, 320, Wrt_bf, 320,
                                                    capred, 128, b_reduce_txt, 320);
    cvt_pad_kernel<<<g1(2048L * 128), 256, 0, stream>>>(capred, capR_bf, 2048, 128, 128);

    // ---- generated weights: (49152 x 64) = gen_W @ bases^T ----
    gemm_bf16_wmma<<<dim3(3072, 1), 32, 0, stream>>>(genWih_bf, 128, bases_bf, 128,
                                                     genC, 64, (const float*)nullptr, 128);
    gen_convert_kernel<<<g1(49152L * 64), 256, 0, stream>>>(genC, gen_bih, WihGen_bf);
    gemm_bf16_wmma<<<dim3(3072, 1), 32, 0, stream>>>(genWhh_bf, 128, bases_bf, 128,
                                                     genC, 64, (const float*)nullptr, 128);
    gen_convert_kernel<<<g1(49152L * 64), 256, 0, stream>>>(genC, gen_bhh, WhhGen_bf);
    gen_bias_kernel<<<g1(64L * 384), 256, 0, stream>>>(gen_Wbih, gen_bbih, igvec, bihgen);
    gen_bias_kernel<<<g1(64L * 384), 256, 0, stream>>>(gen_Wbhh, gen_bbhh, igvec, bhhgen);

    // ---- fused per-image generated GRU (persistent LDS, 240KB) ----
    size_t lds = GG_ACC_BYTES + GG_WHH_BYTES + 64 * 128 * 2;
    gen_gru_kernel<<<64, 256, lds, stream>>>(capR_bf, WihGen_bf, WhhGen_bf,
                                             bihgen, bhhgen, txtvec_bf);

    // ---- final FC + cosine sims ----
    gemm_bf16_wmma<<<dim3(4, 16), 32, 0, stream>>>(te_bf, 1024, Wfc_bf, 1152,
                                                   basefc, 1024, b_txt_fc, 1024);
    fc_batched_wmma<<<dim3(4, 16, 64), 32, 0, stream>>>(txtvec_bf, Wfc_bf + 1024,
                                                        basefc, fcout);
    sims_kernel<<<dim3(64, 64), 256, 0, stream>>>(fcout, iv, sims);
}
